// StaffFasterRCNN_90683939487835
// MI455X (gfx1250) — compile-verified, matrix operand order checked by reading, not run on hardware
//
#include <hip/hip_runtime.h>
#include <math.h>

// ---------------------------------------------------------------------------
// Types for CDNA5 WMMA (wave32): v_wmma_f32_16x16x32_bf16
// A fragment layout (per ISA 7.12.2, 16-bit A 16x32): lane l -> row M=(l&15),
// K chunks [koff..koff+7] and [16+koff..16+koff+7], koff=(l>>4)*8.
// B is stored transposed [N][K] and loaded with the mirrored mapping
// (lane -> column N=(l&15), same K chunking).  C/D: VGPR r, lane l ->
// (M = r + 8*(l>>4), N = (l&15)) within the 16x16 tile.
// ---------------------------------------------------------------------------
typedef __attribute__((ext_vector_type(16))) __bf16 v16bf;
typedef __attribute__((ext_vector_type(8)))  float  v8f;
typedef __attribute__((ext_vector_type(8)))  unsigned short ushort8;

#define DEVFN static __device__ __forceinline__

DEVFN unsigned short f2bf(float f) {
  unsigned u = __float_as_uint(f);
  u += 0x7FFFu + ((u >> 16) & 1u);          // round-to-nearest-even
  return (unsigned short)(u >> 16);
}
DEVFN float bf2f(unsigned short h) { return __uint_as_float(((unsigned)h) << 16); }

union FragU { v16bf v; ushort8 h[2]; };
DEVFN v16bf mkfrag(ushort8 a, ushort8 b) { FragU f; f.h[0] = a; f.h[1] = b; return f.v; }
DEVFN v8f vzero() { v8f z; for (int i = 0; i < 8; ++i) z[i] = 0.f; return z; }

#define WMMA_BF16(A, B, C) \
  __builtin_amdgcn_wmma_f32_16x16x32_bf16(false, (A), false, (B), (short)0, (C), false, false)

#define LOG_CLAMP 4.135166556742356f
#define W_IMG 640.f
#define H_IMG 832.f

// ---------------------------------------------------------------------------
// Conversion kernels
// ---------------------------------------------------------------------------
__global__ void k_pad_nhwc(const float* __restrict__ f, unsigned short* __restrict__ xp,
                           int H, int W) {
  int Hp = H + 2, Wp = W + 2;
  size_t total = (size_t)2 * Hp * Wp * 256;
  size_t stride = (size_t)gridDim.x * blockDim.x;
  for (size_t i = (size_t)blockIdx.x * blockDim.x + threadIdx.x; i < total; i += stride) {
    int c = (int)(i % 256); size_t t = i / 256;
    int px = (int)(t % Wp); t /= Wp;
    int py = (int)(t % Hp); int b = (int)(t / Hp);
    float v = 0.f;
    if (py >= 1 && py <= H && px >= 1 && px <= W)
      v = f[(((size_t)b * 256 + c) * H + (py - 1)) * W + (px - 1)];
    xp[i] = f2bf(v);
  }
}

__global__ void k_wconv(const float* __restrict__ w, unsigned short* __restrict__ o) {
  int i = blockIdx.x * blockDim.x + threadIdx.x;
  if (i >= 256 * 2304) return;
  int n = i / 2304, k = i % 2304;
  int t = k / 256, c = k % 256, ky = t / 3, kx = t % 3;
  o[i] = f2bf(w[((n * 256 + c) * 3 + ky) * 3 + kx]);
}

__global__ void k_w1x1(const float* __restrict__ w, unsigned short* __restrict__ o,
                       int rows, int rowsPad) {
  int i = blockIdx.x * blockDim.x + threadIdx.x;
  if (i >= rowsPad * 256) return;
  int r = i / 256, c = i % 256;
  o[i] = (r < rows) ? f2bf(w[r * 256 + c]) : (unsigned short)0;
}

__global__ void k_wT(const float* __restrict__ w, unsigned short* __restrict__ o,
                     int K, int N) {
  size_t i = (size_t)blockIdx.x * blockDim.x + threadIdx.x;
  size_t total = (size_t)K * N;
  if (i >= total) return;
  size_t n = i / K, k = i % K;
  o[i] = f2bf(w[k * N + n]);
}

// ---------------------------------------------------------------------------
// Fused RPN: 3x3 conv (implicit GEMM, K = 9*256) + ReLU + 1x1 cls/reg heads.
// Workgroup = 4 waves (4 adjacent output rows, same 16-pixel x-tile).
// Per K-step the block stages the 16 KB weight slab (all 256 output channels,
// 32 K) into LDS via async global->LDS (ASYNCcnt, double buffered); each wave
// then runs 16 WMMAs consuming B from LDS and its private A row from global.
// The hidden 16x256 tile goes to LDS (aliasing the dead staging buffers) and
// feeds the 1x1 cls/reg head WMMAs.
// ---------------------------------------------------------------------------
__global__ __launch_bounds__(128) void k_rpn_fused(
    const unsigned short* __restrict__ xpad,   // [2][H+2][W+2][256] bf16
    const unsigned short* __restrict__ wB,     // [256][2304] bf16 (n-major, tap*256+c)
    const unsigned short* __restrict__ wCls,   // [32][256] bf16 (rows >=20 zero)
    const unsigned short* __restrict__ wReg,   // [80][256] bf16
    const float* __restrict__ bConv, const float* __restrict__ bCls,
    const float* __restrict__ bReg,
    float* __restrict__ objs,                  // [2][H*W*20]
    float* __restrict__ dels,                  // [2][H*W*80]
    int H, int W) {
  const int tilesPerRow = (W + 15) >> 4;
  const int rowGroups = (H + 3) >> 2;
  int t = blockIdx.x;
  int b = t / (rowGroups * tilesPerRow);
  int r = t - b * rowGroups * tilesPerRow;
  int yg = r / tilesPerRow;
  int x0 = (r - yg * tilesPerRow) << 4;
  const int tid = threadIdx.x;
  const int w = tid >> 5;            // wave id: output row within group
  const int lane = tid & 31;
  const int mrow = lane & 15;
  const int koff = (lane >> 4) << 3;
  const int Wp = W + 2;
  int yw = yg * 4 + w;
  int ycl = yw < H ? yw : H - 1;     // clamp loads at bottom edge
  bool yok = yw < H;
  int xc = x0 + mrow; if (xc > W - 1) xc = W - 1;   // clamp loads at right edge

  // 36 KB shared region: [2][256][32] bf16 staging (32 KB) aliased later by
  // per-wave hidden tiles [4][16][272] bf16 (34 KB).
  __shared__ __align__(16) unsigned char smemraw[36864];
  unsigned short (*stage)[256][32] = (unsigned short (*)[256][32])smemraw;
  unsigned short (*hid)[16][272]   = (unsigned short (*)[16][272])smemraw;

  // Stage one 16 KB weight slab (K-step s: tap=s/8, c0=(s%8)*32) into LDS
  // with 8 async b128 ops per thread-block iteration.
  auto stage_step = [&](int s, int buf) {
    int tap = s >> 3, c0 = (s & 7) << 5;
    const unsigned short* src = wB + (size_t)(tap * 256 + c0);
    for (int i = 0; i < 8; ++i) {
      int j = i * 128 + tid;                 // 1024 chunks of 16 B
      int n = j >> 2, ch = j & 3;
      const unsigned short* g = src + (size_t)n * 2304 + ch * 8;
      unsigned lds = (unsigned)(size_t)&stage[buf][n][ch * 8];
      asm volatile("global_load_async_to_lds_b128 %0, %1, off"
                   :: "v"(lds), "v"((unsigned long long)(size_t)g)
                   : "memory");
    }
  };

  v8f acc[16];
  for (int i = 0; i < 16; ++i) acc[i] = vzero();

  stage_step(0, 0);
  for (int s = 0; s < 72; ++s) {
    if (s + 1 < 72) {
      stage_step(s + 1, (s + 1) & 1);
      asm volatile("s_wait_asynccnt 8" ::: "memory");   // current slab landed
    } else {
      asm volatile("s_wait_asynccnt 0" ::: "memory");
    }
    __syncthreads();
    int tap = s >> 3, ky = tap / 3, kx = tap % 3, c0 = (s & 7) << 5;
    const unsigned short* arow =
        xpad + (((size_t)b * (H + 2) + (size_t)(ycl + ky)) * Wp + (size_t)(xc + kx)) * 256;
    v16bf fa = mkfrag(*(const ushort8*)(arow + c0 + koff),
                      *(const ushort8*)(arow + c0 + 16 + koff));
    const unsigned short* sb = &stage[s & 1][0][0];
#pragma unroll
    for (int nt = 0; nt < 16; ++nt) {
      const unsigned short* brow = sb + (nt * 16 + mrow) * 32;
      v16bf fb = mkfrag(*(const ushort8*)(brow + koff),
                        *(const ushort8*)(brow + 16 + koff));
      acc[nt] = WMMA_BF16(fa, fb, acc[nt]);
    }
    __syncthreads();   // all waves done with this buffer before re-staging
  }

  // bias + ReLU -> per-wave hidden tile in LDS (aliases dead staging buffers)
  const int mhalf = (lane >> 4) << 3;
#pragma unroll
  for (int nt = 0; nt < 16; ++nt) {
    int n = nt * 16 + mrow;
    float bias = bConv[n];
#pragma unroll
    for (int rr = 0; rr < 8; ++rr) {
      float v = acc[nt][rr] + bias;
      hid[w][mhalf + rr][n] = f2bf(v > 0.f ? v : 0.f);
    }
  }
  __syncthreads();

  v8f oa[2], ra[5];
  for (int i = 0; i < 2; ++i) oa[i] = vzero();
  for (int i = 0; i < 5; ++i) ra[i] = vzero();
  for (int cs = 0; cs < 8; ++cs) {
    const int c0 = cs << 5;
    v16bf fa = mkfrag(*(const ushort8*)(&hid[w][mrow][c0 + koff]),
                      *(const ushort8*)(&hid[w][mrow][c0 + 16 + koff]));
#pragma unroll
    for (int nt = 0; nt < 2; ++nt) {
      const unsigned short* brow = wCls + (size_t)(nt * 16 + mrow) * 256 + c0;
      v16bf fb = mkfrag(*(const ushort8*)(brow + koff),
                        *(const ushort8*)(brow + 16 + koff));
      oa[nt] = WMMA_BF16(fa, fb, oa[nt]);
    }
#pragma unroll
    for (int nt = 0; nt < 5; ++nt) {
      const unsigned short* brow = wReg + (size_t)(nt * 16 + mrow) * 256 + c0;
      v16bf fb = mkfrag(*(const ushort8*)(brow + koff),
                        *(const ushort8*)(brow + 16 + koff));
      ra[nt] = WMMA_BF16(fa, fb, ra[nt]);
    }
  }

  const size_t HW = (size_t)H * W;
  float* objB = objs + (size_t)b * HW * 20;
  float* delB = dels + (size_t)b * HW * 80;
#pragma unroll
  for (int nt = 0; nt < 2; ++nt) {
    int n = nt * 16 + mrow;
    if (n < 20) {
      float bias = bCls[n];
#pragma unroll
      for (int rr = 0; rr < 8; ++rr) {
        int xx = x0 + mhalf + rr;
        if (yok && xx < W) objB[((size_t)yw * W + xx) * 20 + n] = oa[nt][rr] + bias;
      }
    }
  }
#pragma unroll
  for (int nt = 0; nt < 5; ++nt) {
    int n = nt * 16 + mrow;
    float bias = bReg[n];
#pragma unroll
    for (int rr = 0; rr < 8; ++rr) {
      int xx = x0 + mhalf + rr;
      if (yok && xx < W) delB[((size_t)yw * W + xx) * 80 + n] = ra[nt][rr] + bias;
    }
  }
}

// ---------------------------------------------------------------------------
// top-k(100) per (image, level): chunk-max array + iterative extract
// ---------------------------------------------------------------------------
__global__ void k_chunkmax(const float* __restrict__ objs, float* __restrict__ cmax,
                           int N, int nchunks) {
  int i = blockIdx.x * blockDim.x + threadIdx.x;
  if (i >= 2 * nchunks) return;
  int b = i / nchunks, ch = i % nchunks;
  const float* p = objs + (size_t)b * N;
  int s = ch * 64, e = s + 64; if (e > N) e = N;
  float m = -INFINITY;
  for (int j = s; j < e; ++j) m = fmaxf(m, p[j]);
  cmax[i] = m;
}

__global__ __launch_bounds__(256) void k_topk(
    float* __restrict__ objs, float* __restrict__ cmax,
    float* __restrict__ tv, int* __restrict__ ti, int N, int nchunks, int lvl) {
  int b = blockIdx.x, tid = threadIdx.x;
  float* o  = objs + (size_t)b * N;
  float* cm = cmax + (size_t)b * nchunks;
  float* tvb = tv + (size_t)(b * 5 + lvl) * 100;
  int*   tib = ti + (size_t)(b * 5 + lvl) * 100;
  __shared__ float sv[256];
  __shared__ int   si[256];
  __shared__ int   bestChunk;
  for (int it = 0; it < 100; ++it) {
    float mv = -INFINITY; int mi = 0;
    for (int c = tid; c < nchunks; c += 256) {
      float v = cm[c];
      if (v > mv) { mv = v; mi = c; }
    }
    sv[tid] = mv; si[tid] = mi; __syncthreads();
    for (int s = 128; s > 0; s >>= 1) {
      if (tid < s && sv[tid + s] > sv[tid]) { sv[tid] = sv[tid + s]; si[tid] = si[tid + s]; }
      __syncthreads();
    }
    if (tid == 0) bestChunk = si[0];
    __syncthreads();
    int c = bestChunk, s0 = c * 64, e0 = s0 + 64; if (e0 > N) e0 = N;
    float ev = -INFINITY; int ei = s0;
    if (tid < 64) { int j = s0 + tid; if (j < e0) { ev = o[j]; ei = j; } }
    sv[tid] = ev; si[tid] = ei; __syncthreads();
    for (int s = 32; s > 0; s >>= 1) {
      if (tid < s && sv[tid + s] > sv[tid]) { sv[tid] = sv[tid + s]; si[tid] = si[tid + s]; }
      __syncthreads();
    }
    if (tid == 0) { tvb[it] = sv[0]; tib[it] = si[0]; o[si[0]] = -INFINITY; }
    __syncthreads();
    float nv = -INFINITY;
    if (tid < 64) { int j = s0 + tid; if (j < e0) nv = o[j]; }
    sv[tid] = nv; __syncthreads();
    for (int s = 32; s > 0; s >>= 1) {
      if (tid < s) sv[tid] = fmaxf(sv[tid], sv[tid + s]);
      __syncthreads();
    }
    if (tid == 0) cm[c] = sv[0];
    __syncthreads();
  }
}

// ---------------------------------------------------------------------------
// anchor decode + clip + sigmoid for the 100 candidates of one level
// ---------------------------------------------------------------------------
__global__ void k_decode(const float* __restrict__ tv, const int* __restrict__ ti,
                         const float* __restrict__ dels,
                         float* __restrict__ cboxes, float* __restrict__ cscores,
                         int W, int strideLvl, int lvl, size_t delImgStride) {
  int t = blockIdx.x * blockDim.x + threadIdx.x;
  if (t >= 200) return;
  int b = t / 100, rank = t % 100;
  int slot = b * 500 + lvl * 100 + rank;
  float logit = tv[(b * 5 + lvl) * 100 + rank];
  int idx = ti[(b * 5 + lvl) * 100 + rank];
  int p = idx / 20, a = idx % 20;
  int yy = p / W, xx = p % W;
  const float SIZES[5]  = {64.f, 128.f, 256.f, 512.f, 1024.f};
  const float RATIOS[4] = {0.04f, 0.05f, 0.08f, 0.15f};
  float scale = SIZES[a % 5];
  float ratio = RATIOS[a / 5];
  float hr = sqrtf(ratio), wr = 1.f / hr;
  float hw = rintf(wr * scale * 0.5f);
  float hh = rintf(hr * scale * 0.5f);
  float sx = (float)(xx * strideLvl), sy = (float)(yy * strideLvl);
  float ax1 = sx - hw, ay1 = sy - hh, ax2 = sx + hw, ay2 = sy + hh;
  const float* d = dels + (size_t)b * delImgStride + (size_t)p * 80 + a * 4;
  float wdt = ax2 - ax1, hgt = ay2 - ay1;
  float cx = ax1 + 0.5f * wdt, cy = ay1 + 0.5f * hgt;
  float dw = fminf(d[2], LOG_CLAMP), dh = fminf(d[3], LOG_CLAMP);
  float pcx = d[0] * wdt + cx, pcy = d[1] * hgt + cy;
  float pw = expf(dw) * wdt, ph = expf(dh) * hgt;
  float x1 = fminf(fmaxf(pcx - 0.5f * pw, 0.f), W_IMG);
  float x2 = fminf(fmaxf(pcx + 0.5f * pw, 0.f), W_IMG);
  float y1 = fminf(fmaxf(pcy - 0.5f * ph, 0.f), H_IMG);
  float y2 = fminf(fmaxf(pcy + 0.5f * ph, 0.f), H_IMG);
  bool valid = (x2 - x1 >= 1e-3f) && (y2 - y1 >= 1e-3f);
  float sc = 1.f / (1.f + expf(-logit));
  cboxes[slot * 4 + 0] = x1; cboxes[slot * 4 + 1] = y1;
  cboxes[slot * 4 + 2] = x2; cboxes[slot * 4 + 3] = y2;
  cscores[slot] = valid ? sc : -INFINITY;
}

// ---------------------------------------------------------------------------
// RPN NMS: bitonic sort (512) + greedy suppression + top-80 selection
// ---------------------------------------------------------------------------
__global__ __launch_bounds__(512) void k_rpn_nms(
    const float* __restrict__ cboxes, const float* __restrict__ cscores,
    float* __restrict__ props, float* __restrict__ pscores) {
  int b = blockIdx.x, tid = threadIdx.x;
  __shared__ float bx[512][4];   // level-offset boxes (for IoU), original index
  __shared__ float obx[512][4];  // clipped boxes (for output), original index
  __shared__ float area[512];
  __shared__ float sv[512];
  __shared__ short sid[512];
  __shared__ int   supp[512];
  __shared__ short pick[80];
  if (tid < 500) {
    float x1 = cboxes[(b * 500 + tid) * 4 + 0];
    float y1 = cboxes[(b * 500 + tid) * 4 + 1];
    float x2 = cboxes[(b * 500 + tid) * 4 + 2];
    float y2 = cboxes[(b * 500 + tid) * 4 + 3];
    obx[tid][0] = x1; obx[tid][1] = y1; obx[tid][2] = x2; obx[tid][3] = y2;
    float off = (float)(tid / 100) * 834.f;   // per-level batched NMS offset
    bx[tid][0] = x1 + off; bx[tid][1] = y1 + off;
    bx[tid][2] = x2 + off; bx[tid][3] = y2 + off;
    area[tid] = (x2 - x1) * (y2 - y1);
    sv[tid] = cscores[b * 500 + tid];
  } else {
    obx[tid][0] = obx[tid][1] = obx[tid][2] = obx[tid][3] = 0.f;
    bx[tid][0] = bx[tid][1] = bx[tid][2] = bx[tid][3] = 0.f;
    area[tid] = 0.f; sv[tid] = -INFINITY;
  }
  sid[tid] = (short)tid; supp[tid] = 0;
  __syncthreads();
  for (int k = 2; k <= 512; k <<= 1)
    for (int j = k >> 1; j > 0; j >>= 1) {
      int ixj = tid ^ j;
      if (ixj > tid) {
        bool desc = ((tid & k) == 0);
        float a = sv[tid], c = sv[ixj];
        if (desc ? (a < c) : (a > c)) {
          sv[tid] = c; sv[ixj] = a;
          short t0 = sid[tid]; sid[tid] = sid[ixj]; sid[ixj] = t0;
        }
      }
      __syncthreads();
    }
  for (int i = 0; i < 500; ++i) {
    if (!supp[i] && tid > i) {
      int oi = sid[i], oj = sid[tid];
      float xx1 = fmaxf(bx[oi][0], bx[oj][0]);
      float yy1 = fmaxf(bx[oi][1], bx[oj][1]);
      float xx2 = fminf(bx[oi][2], bx[oj][2]);
      float yy2 = fminf(bx[oi][3], bx[oj][3]);
      float inter = fmaxf(xx2 - xx1, 0.f) * fmaxf(yy2 - yy1, 0.f);
      float iou = inter / (area[oi] + area[oj] - inter + 1e-9f);
      if (iou > 0.7f) supp[tid] = 1;
    }
    __syncthreads();
  }
  if (tid == 0) {
    int cnt = 0;
    for (int i = 0; i < 500 && cnt < 80; ++i) if (!supp[i]) pick[cnt++] = (short)i;
    for (int i = 0; i < 500 && cnt < 80; ++i) if (supp[i])  pick[cnt++] = (short)i;
  }
  __syncthreads();
  if (tid < 80) {
    int p = pick[tid], orig = sid[p];
    pscores[b * 80 + tid] = supp[p] ? -INFINITY : sv[p];
    props[(b * 80 + tid) * 4 + 0] = obx[orig][0];
    props[(b * 80 + tid) * 4 + 1] = obx[orig][1];
    props[(b * 80 + tid) * 4 + 2] = obx[orig][2];
    props[(b * 80 + tid) * 4 + 3] = obx[orig][3];
  }
}

// ---------------------------------------------------------------------------
// ROI align (FPN level assignment, 7x7, sampling_ratio 2) -> bf16 rows [K=12544]
// ---------------------------------------------------------------------------
__global__ __launch_bounds__(128) void k_roialign(
    const float* __restrict__ f0, const float* __restrict__ f1,
    const float* __restrict__ f2, const float* __restrict__ f3,
    const float* __restrict__ props, unsigned short* __restrict__ xpool) {
  int roi = blockIdx.x;               // 0..159
  int b = roi / 80;
  int tid = threadIdx.x;
  __shared__ int   li;
  __shared__ float sxs[14], sys[14];
  __shared__ int   sy0[196], sx0[196], sy1[196], sx1[196];
  __shared__ float sly[196], slx[196], sok[196];
  float x1 = props[roi * 4 + 0], y1 = props[roi * 4 + 1];
  float x2 = props[roi * 4 + 2], y2 = props[roi * 4 + 3];
  if (tid == 0) {
    float w = fmaxf(x2 - x1, 0.f), h = fmaxf(y2 - y1, 0.f);
    float lv = floorf(4.f + log2f(sqrtf(w * h) / 224.f + 1e-6f));
    lv = fminf(fmaxf(lv, 2.f), 5.f);
    li = (int)lv - 2;
  }
  __syncthreads();
  const int HL4[4] = {208, 104, 52, 26};
  const int WL4[4] = {160, 80, 40, 20};
  const float SC4[4] = {0.25f, 0.125f, 0.0625f, 0.03125f};
  int l = li, FH = HL4[l], FW = WL4[l];
  float sc = SC4[l];
  const float* fp = (l == 0 ? f0 : l == 1 ? f1 : l == 2 ? f2 : f3) +
                    (size_t)b * 256 * FH * FW;
  float rx1 = x1 * sc, ry1 = y1 * sc;
  float rw = fmaxf(x2 * sc - rx1, 1.f), rh = fmaxf(y2 * sc - ry1, 1.f);
  if (tid < 14) { sxs[tid] = rx1 + (tid + 0.5f) * (rw / 14.f);
                  sys[tid] = ry1 + (tid + 0.5f) * (rh / 14.f); }
  __syncthreads();
  for (int s = tid; s < 196; s += blockDim.x) {
    int gy = s / 14, gx = s % 14;
    float yv = sys[gy], xv = sxs[gx];
    sok[s] = (yv > -1.f && yv < (float)FH && xv > -1.f && xv < (float)FW) ? 1.f : 0.f;
    float yc = fminf(fmaxf(yv, 0.f), (float)FH - 1.f);
    float xc = fminf(fmaxf(xv, 0.f), (float)FW - 1.f);
    int y0 = (int)floorf(yc), x0 = (int)floorf(xc);
    sy0[s] = y0; sx0[s] = x0;
    sy1[s] = (y0 + 1 < FH) ? y0 + 1 : FH - 1;
    sx1[s] = (x0 + 1 < FW) ? x0 + 1 : FW - 1;
    sly[s] = yc - (float)y0; slx[s] = xc - (float)x0;
  }
  __syncthreads();
  for (int c = tid; c < 256; c += blockDim.x) {
    const float* fc = fp + (size_t)c * FH * FW;
    for (int py = 0; py < 7; ++py)
      for (int px = 0; px < 7; ++px) {
        float accv = 0.f;
        for (int sy = 0; sy < 2; ++sy)
          for (int sx = 0; sx < 2; ++sx) {
            int s = (py * 2 + sy) * 14 + (px * 2 + sx);
            float ly = sly[s], lx = slx[s];
            float v00 = fc[sy0[s] * FW + sx0[s]], v01 = fc[sy0[s] * FW + sx1[s]];
            float v10 = fc[sy1[s] * FW + sx0[s]], v11 = fc[sy1[s] * FW + sx1[s]];
            accv += (v00 * (1 - ly) * (1 - lx) + v01 * (1 - ly) * lx +
                     v10 * ly * (1 - lx) + v11 * ly * lx) * sok[s];
          }
        xpool[(size_t)roi * 12544 + (size_t)c * 49 + py * 7 + px] = f2bf(accv * 0.25f);
      }
  }
}

// ---------------------------------------------------------------------------
// Generic bf16 WMMA GEMM: Out[M][N] = relu(A[M][K] * Bt[N][K]^T + bias)
// One wave: 16 rows x 128 cols (8 accumulators).
// ---------------------------------------------------------------------------
__global__ __launch_bounds__(32) void k_gemm(
    const unsigned short* __restrict__ A, const unsigned short* __restrict__ Bt,
    const float* __restrict__ bias, unsigned short* __restrict__ Out,
    int N, int K, int relu) {
  int mt = blockIdx.x, n0 = blockIdx.y * 128;
  int lane = threadIdx.x, mrow = lane & 15, koff = (lane >> 4) << 3;
  const unsigned short* arow = A + (size_t)(mt * 16 + mrow) * K;
  v8f acc[8];
  for (int i = 0; i < 8; ++i) acc[i] = vzero();
  int ksteps = K >> 5;
  for (int ks = 0; ks < ksteps; ++ks) {
    int kb = ks << 5;
    __builtin_prefetch(arow + kb + 128, 0, 1);   // global_prefetch_b8 ahead on A
    v16bf fa = mkfrag(*(const ushort8*)(arow + kb + koff),
                      *(const ushort8*)(arow + kb + 16 + koff));
#pragma unroll
    for (int nt = 0; nt < 8; ++nt) {
      const unsigned short* brow = Bt + (size_t)(n0 + nt * 16 + mrow) * K + kb;
      v16bf fb = mkfrag(*(const ushort8*)(brow + koff),
                        *(const ushort8*)(brow + 16 + koff));
      acc[nt] = WMMA_BF16(fa, fb, acc[nt]);
    }
  }
  int mhalf = (lane >> 4) << 3;
#pragma unroll
  for (int nt = 0; nt < 8; ++nt) {
    int n = n0 + nt * 16 + mrow;
    float bs = bias[n];
#pragma unroll
    for (int rr = 0; rr < 8; ++rr) {
      float v = acc[nt][rr] + bs;
      if (relu) v = v > 0.f ? v : 0.f;
      Out[(size_t)(mt * 16 + mhalf + rr) * N + n] = f2bf(v);
    }
  }
}

// Small heads: logits [160][2], deltas [160][8]
__global__ void k_heads(const unsigned short* __restrict__ h2,
                        const float* __restrict__ cw, const float* __restrict__ cb,
                        const float* __restrict__ rw, const float* __restrict__ rb,
                        float* __restrict__ logits, float* __restrict__ del8) {
  int t = blockIdx.x * blockDim.x + threadIdx.x;
  if (t >= 160 * 10) return;
  int row = t / 10, j = t % 10;
  const unsigned short* h = h2 + (size_t)row * 1024;
  float s = 0.f;
  if (j < 2) {
    for (int k = 0; k < 1024; ++k) s += bf2f(h[k]) * cw[k * 2 + j];
    logits[row * 2 + j] = s + cb[j];
  } else {
    int jj = j - 2;
    for (int k = 0; k < 1024; ++k) s += bf2f(h[k]) * rw[k * 8 + jj];
    del8[row * 8 + jj] = s + rb[jj];
  }
}

// ---------------------------------------------------------------------------
// Final: softmax score, box decode (BBOX_W), clip, validity, NMS(0.5), output
// ---------------------------------------------------------------------------
__global__ __launch_bounds__(128) void k_final(
    const float* __restrict__ props, const float* __restrict__ pscores,
    const float* __restrict__ logits, const float* __restrict__ del8,
    float* __restrict__ outBoxes, float* __restrict__ outScores) {
  int b = blockIdx.x, tid = threadIdx.x;
  __shared__ float bx[128][4];
  __shared__ float sv[128];
  __shared__ short sid[128];
  __shared__ int   supp[128];
  __shared__ int   validA[128];
  __shared__ float rawsc[128];
  __shared__ int   keepm[128];
  if (tid < 80) {
    int row = b * 80 + tid;
    float l0 = logits[row * 2], l1 = logits[row * 2 + 1];
    float sc = 1.f / (1.f + expf(l0 - l1));
    float px1 = props[row * 4 + 0], py1 = props[row * 4 + 1];
    float px2 = props[row * 4 + 2], py2 = props[row * 4 + 3];
    float wdt = px2 - px1, hgt = py2 - py1;
    float cx = px1 + 0.5f * wdt, cy = py1 + 0.5f * hgt;
    float dx = del8[row * 8 + 4] * 0.1f, dy = del8[row * 8 + 5] * 0.1f;
    float dw = fminf(del8[row * 8 + 6] * 0.2f, LOG_CLAMP);
    float dh = fminf(del8[row * 8 + 7] * 0.2f, LOG_CLAMP);
    float pcx = dx * wdt + cx, pcy = dy * hgt + cy;
    float pw = expf(dw) * wdt, ph = expf(dh) * hgt;
    float x1 = fminf(fmaxf(pcx - 0.5f * pw, 0.f), W_IMG);
    float x2 = fminf(fmaxf(pcx + 0.5f * pw, 0.f), W_IMG);
    float y1 = fminf(fmaxf(pcy - 0.5f * ph, 0.f), H_IMG);
    float y2 = fminf(fmaxf(pcy + 0.5f * ph, 0.f), H_IMG);
    float ps = pscores[row];
    bool val = (ps != -INFINITY) && (x2 - x1 >= 1e-2f) && (y2 - y1 >= 1e-2f) &&
               (sc > 0.05f);
    bx[tid][0] = x1; bx[tid][1] = y1; bx[tid][2] = x2; bx[tid][3] = y2;
    rawsc[tid] = sc; validA[tid] = val ? 1 : 0;
    sv[tid] = val ? sc : -INFINITY;
  } else {
    bx[tid][0] = bx[tid][1] = bx[tid][2] = bx[tid][3] = 0.f;
    rawsc[tid] = 0.f; validA[tid] = 0; sv[tid] = -INFINITY;
  }
  sid[tid] = (short)tid; supp[tid] = 0;
  __syncthreads();
  for (int k = 2; k <= 128; k <<= 1)
    for (int j = k >> 1; j > 0; j >>= 1) {
      int ixj = tid ^ j;
      if (ixj > tid) {
        bool desc = ((tid & k) == 0);
        float a = sv[tid], c = sv[ixj];
        if (desc ? (a < c) : (a > c)) {
          sv[tid] = c; sv[ixj] = a;
          short t0 = sid[tid]; sid[tid] = sid[ixj]; sid[ixj] = t0;
        }
      }
      __syncthreads();
    }
  for (int i = 0; i < 80; ++i) {
    if (!supp[i] && tid > i) {
      int oi = sid[i], oj = sid[tid];
      float ai = (bx[oi][2] - bx[oi][0]) * (bx[oi][3] - bx[oi][1]);
      float aj = (bx[oj][2] - bx[oj][0]) * (bx[oj][3] - bx[oj][1]);
      float xx1 = fmaxf(bx[oi][0], bx[oj][0]);
      float yy1 = fmaxf(bx[oi][1], bx[oj][1]);
      float xx2 = fminf(bx[oi][2], bx[oj][2]);
      float yy2 = fminf(bx[oi][3], bx[oj][3]);
      float inter = fmaxf(xx2 - xx1, 0.f) * fmaxf(yy2 - yy1, 0.f);
      float iou = inter / (ai + aj - inter + 1e-9f);
      if (iou > 0.5f) supp[tid] = 1;
    }
    __syncthreads();
  }
  keepm[sid[tid]] = (!supp[tid]) && validA[sid[tid]];
  __syncthreads();
  if (tid < 80) {
    int row = b * 80 + tid;
    bool kp = keepm[tid];
    outScores[row] = kp ? rawsc[tid] : 0.f;
    for (int q = 0; q < 4; ++q) outBoxes[row * 4 + q] = kp ? bx[tid][q] : 0.f;
  }
}

// ---------------------------------------------------------------------------
// Host: workspace carving + launch sequence (all on `stream`, graph-capturable)
// ---------------------------------------------------------------------------
extern "C" void kernel_launch(void* const* d_in, const int* in_sizes, int n_in,
                              void* d_out, int out_size, void* d_ws, size_t ws_size,
                              hipStream_t stream) {
  (void)in_sizes; (void)n_in; (void)out_size; (void)ws_size;
  static const int HL[5] = {208, 104, 52, 26, 13};
  static const int WL[5] = {160, 80, 40, 20, 10};
  static const int STRL[5] = {4, 8, 16, 32, 64};

  const float* feats[5];
  for (int i = 0; i < 5; ++i) feats[i] = (const float*)d_in[i];
  const float* rpn_conv_w = (const float*)d_in[5];
  const float* rpn_conv_b = (const float*)d_in[6];
  const float* rpn_cls_w  = (const float*)d_in[7];
  const float* rpn_cls_b  = (const float*)d_in[8];
  const float* rpn_reg_w  = (const float*)d_in[9];
  const float* rpn_reg_b  = (const float*)d_in[10];
  const float* fc1_w = (const float*)d_in[11];
  const float* fc1_b = (const float*)d_in[12];
  const float* fc2_w = (const float*)d_in[13];
  const float* fc2_b = (const float*)d_in[14];
  const float* cls_w = (const float*)d_in[15];
  const float* cls_b = (const float*)d_in[16];
  const float* reg_w = (const float*)d_in[17];
  const float* reg_b = (const float*)d_in[18];

  unsigned char* wsb = (unsigned char*)d_ws;
  size_t cur = 0;
  auto alloc = [&](size_t bytes) -> void* {
    void* p = wsb + cur;
    cur = (cur + bytes + 255) & ~(size_t)255;
    return p;
  };

  unsigned short* xpad[5];
  for (int l = 0; l < 5; ++l)
    xpad[l] = (unsigned short*)alloc((size_t)2 * (HL[l] + 2) * (WL[l] + 2) * 256 * 2);
  unsigned short* wConv = (unsigned short*)alloc((size_t)256 * 2304 * 2);
  unsigned short* wCls  = (unsigned short*)alloc((size_t)32 * 256 * 2);
  unsigned short* wReg  = (unsigned short*)alloc((size_t)80 * 256 * 2);
  float* objs[5]; float* dels[5]; float* cmax[5]; int nch[5]; int HWl[5];
  for (int l = 0; l < 5; ++l) {
    HWl[l] = HL[l] * WL[l];
    objs[l] = (float*)alloc((size_t)2 * HWl[l] * 20 * 4);
    dels[l] = (float*)alloc((size_t)2 * HWl[l] * 80 * 4);
    nch[l] = (HWl[l] * 20 + 63) / 64;
    cmax[l] = (float*)alloc((size_t)2 * nch[l] * 4);
  }
  float* tv = (float*)alloc(1000 * 4);
  int*   ti = (int*)alloc(1000 * 4);
  float* cboxes  = (float*)alloc((size_t)2 * 500 * 4 * 4);
  float* cscores = (float*)alloc((size_t)2 * 500 * 4);
  float* props   = (float*)alloc((size_t)2 * 80 * 4 * 4);
  float* pscores = (float*)alloc((size_t)2 * 80 * 4);
  unsigned short* xpool = (unsigned short*)alloc((size_t)160 * 12544 * 2);
  unsigned short* wtfc1 = (unsigned short*)alloc((size_t)12544 * 1024 * 2);
  unsigned short* wtfc2 = (unsigned short*)alloc((size_t)1024 * 1024 * 2);
  unsigned short* h1 = (unsigned short*)alloc((size_t)160 * 1024 * 2);
  unsigned short* h2 = (unsigned short*)alloc((size_t)160 * 1024 * 2);
  float* logits = (float*)alloc((size_t)160 * 2 * 4);
  float* del8   = (float*)alloc((size_t)160 * 8 * 4);

  // 1) feature NHWC+halo bf16, weight conversions
  for (int l = 0; l < 5; ++l) {
    size_t total = (size_t)2 * (HL[l] + 2) * (WL[l] + 2) * 256;
    int blocks = (int)((total + 255) / 256);
    k_pad_nhwc<<<blocks, 256, 0, stream>>>(feats[l], xpad[l], HL[l], WL[l]);
  }
  k_wconv<<<(256 * 2304 + 255) / 256, 256, 0, stream>>>(rpn_conv_w, wConv);
  k_w1x1<<<(32 * 256 + 255) / 256, 256, 0, stream>>>(rpn_cls_w, wCls, 20, 32);
  k_w1x1<<<(80 * 256 + 255) / 256, 256, 0, stream>>>(rpn_reg_w, wReg, 80, 80);
  k_wT<<<(int)(((size_t)12544 * 1024 + 255) / 256), 256, 0, stream>>>(fc1_w, wtfc1, 12544, 1024);
  k_wT<<<(int)(((size_t)1024 * 1024 + 255) / 256), 256, 0, stream>>>(fc2_w, wtfc2, 1024, 1024);

  // 2) fused RPN conv + heads (WMMA, async-to-LDS weight staging)
  for (int l = 0; l < 5; ++l) {
    int tilesPerRow = (WL[l] + 15) / 16;
    int rowGroups = (HL[l] + 3) / 4;
    int blocks = 2 * rowGroups * tilesPerRow;
    k_rpn_fused<<<blocks, 128, 0, stream>>>(xpad[l], wConv, wCls, wReg,
                                            rpn_conv_b, rpn_cls_b, rpn_reg_b,
                                            objs[l], dels[l], HL[l], WL[l]);
  }

  // 3) per-level top-100 + decode
  for (int l = 0; l < 5; ++l) {
    int N = HWl[l] * 20;
    k_chunkmax<<<(2 * nch[l] + 255) / 256, 256, 0, stream>>>(objs[l], cmax[l], N, nch[l]);
    k_topk<<<2, 256, 0, stream>>>(objs[l], cmax[l], tv, ti, N, nch[l], l);
    k_decode<<<1, 256, 0, stream>>>(tv, ti, dels[l], cboxes, cscores,
                                    WL[l], STRL[l], l, (size_t)HWl[l] * 80);
  }

  // 4) RPN NMS -> 80 proposals per image
  k_rpn_nms<<<2, 512, 0, stream>>>(cboxes, cscores, props, pscores);

  // 5) ROI align -> pooled bf16 rows
  k_roialign<<<160, 128, 0, stream>>>(feats[0], feats[1], feats[2], feats[3],
                                      props, xpool);

  // 6) FC head GEMMs (WMMA): 160x12544x1024, 160x1024x1024
  k_gemm<<<dim3(10, 8), 32, 0, stream>>>(xpool, wtfc1, fc1_b, h1, 1024, 12544, 1);
  k_gemm<<<dim3(10, 8), 32, 0, stream>>>(h1, wtfc2, fc2_b, h2, 1024, 1024, 1);

  // 7) cls/reg heads + final NMS -> output
  k_heads<<<(1600 + 255) / 256, 256, 0, stream>>>(h2, cls_w, cls_b, reg_w, reg_b,
                                                  logits, del8);
  float* outBoxes = (float*)d_out;          // [2,80,4]
  float* outScores = (float*)d_out + 640;   // [2,80]
  k_final<<<2, 128, 0, stream>>>(props, pscores, logits, del8, outBoxes, outScores);
}
